// InfoNCELoss_10110353014828
// MI455X (gfx1250) — matrix operand order repaired and checked by reading, not compiled, not run on hardware
//
#include <hip/hip_runtime.h>
#include <hip/hip_bf16.h>
#include <cstdint>

// ---------------------------------------------------------------------------
// InfoNCE fused kernel for MI455X (gfx1250, wave32, WMMA).
// One workgroup handles one (step, s) slice:
//   scores[b,c] = (X_s @ W[step]^T) @ C_s^T   (64 x 64, K1=1024, K2=1024)
//   loss[b]     = logsumexp_c(scores) - log(B*S) - scores[b,b]
// bf16 WMMA (v_wmma_f32_16x16x32_bf16) with f32 accumulation.
// W is pre-converted to bf16 in d_ws and staged into LDS with
// global_load_async_to_lds_b128 (ASYNCcnt path) when scratch allows.
// ---------------------------------------------------------------------------

typedef __bf16 bf16;
typedef __attribute__((ext_vector_type(16))) __bf16 v16bf;
typedef __attribute__((ext_vector_type(8)))  __bf16 v8bf;
typedef __attribute__((ext_vector_type(8)))  float  v8f;

#define B_    64
#define T_    512
#define DX_   1024
#define DY_   1024

struct __align__(8)  bf16x4 { bf16 a0, a1, a2, a3; };
struct __align__(16) bf16x8 { bf16 a0, a1, a2, a3, a4, a5, a6, a7; };

__device__ __forceinline__ v8f wmma_bf16(v16bf a, v16bf b, v8f c) {
  return __builtin_amdgcn_wmma_f32_16x16x32_bf16(
      /*neg_a=*/false, a, /*neg_b=*/false, b,
      /*c_mod=*/(short)0, c, /*reuse_a=*/false, /*reuse_b=*/false);
}

// Flat shared pointer -> 32-bit LDS byte offset (ISA 10.2: LDS aperture uses addr[31:0]).
__device__ __forceinline__ uint32_t lds_addr32(const void* p) {
  return (uint32_t)(uintptr_t)p;
}

// Async DMA: 16B global -> LDS per lane, tracked by ASYNCcnt (CDNA5 ISA).
__device__ __forceinline__ void async_ld_b128(uint32_t lds_byte_addr, const void* gaddr) {
  unsigned long long ga = (unsigned long long)(uintptr_t)gaddr;
  asm volatile("global_load_async_to_lds_b128 %0, %1, off"
               :: "v"(lds_byte_addr), "v"(ga) : "memory");
}

// A-matrix 16x32 bf16 fragment from row-major LDS tile.
// Lane l: row = row0 + (l&15); K = (l>>4)*8 + {0..7} and +16  (ISA 7.12.2)
__device__ __forceinline__ v16bf load_a_frag(const bf16* base, int row0, int ld, int col0) {
  const int lane = threadIdx.x & 31;
  const bf16* p = base + (row0 + (lane & 15)) * ld + col0 + ((lane >> 4) * 8);
  v8bf lo = *(const v8bf*)(p);
  v8bf hi = *(const v8bf*)(p + 16);
  v16bf a;
#pragma unroll
  for (int i = 0; i < 8; ++i) { a[i] = lo[i]; a[8 + i] = hi[i]; }
  return a;
}

// B-matrix 32x16 bf16 fragment; element (k,n) = base[(row0+n)*ld + col0 + k].
// Lane l: n = l&15; K = (l>>4)*8 + {0..7} and +16 (matches A fragment).
__device__ __forceinline__ v16bf load_b_frag(const bf16* base, int row0, int ld, int col0) {
  const int lane = threadIdx.x & 31;
  const bf16* p = base + (row0 + (lane & 15)) * ld + col0 + ((lane >> 4) * 8);
  v8bf lo = *(const v8bf*)(p);
  v8bf hi = *(const v8bf*)(p + 16);
  v16bf b;
#pragma unroll
  for (int i = 0; i < 8; ++i) { b[i] = lo[i]; b[8 + i] = hi[i]; }
  return b;
}

// ---- pre-pass: W (5 x 1024 x 1024) f32 -> bf16 in workspace ----
__global__ void convert_w_bf16_kernel(const float* __restrict__ Wg,
                                      bf16* __restrict__ Wb) {
  const size_t i = ((size_t)blockIdx.x * blockDim.x + threadIdx.x) * 8;
  float4 a = *(const float4*)(Wg + i);
  float4 b = *(const float4*)(Wg + i + 4);
  bf16x8 o = { (bf16)a.x, (bf16)a.y, (bf16)a.z, (bf16)a.w,
               (bf16)b.x, (bf16)b.y, (bf16)b.z, (bf16)b.w };
  *(bf16x8*)(Wb + i) = o;
}

template <bool ASYNCW>
__global__ void infonce_fused_kernel(const float* __restrict__ emb,
                                     const float* __restrict__ ctx,
                                     const float* __restrict__ Wg,   // f32 W (fallback)
                                     const bf16*  __restrict__ Wb,   // bf16 W in d_ws
                                     float* __restrict__ out)
{
  const int s    = blockIdx.x;
  const int step = blockIdx.y;
  const int S    = T_ - step - 1;            // MIN_CONTEXT = 1
  if (s >= S) return;                        // uniform block-wide exit (pre-barrier)

  // ---- LDS carve-out (dynamic; 311296 B total, fits 320KB/WGP) ----
  extern __shared__ __align__(16) char smem[];
  bf16*  Xb = (bf16*)(smem);                 // 64 x 1024 bf16      (131072 B)
  bf16*  Wt = (bf16*)(smem + 131072);        // 2 x 32 x 1024 bf16  (131072 B)
  bf16*  Ct = (bf16*)(smem + 262144);        // 2 x 64 x 32 bf16    (  8192 B)
  bf16*  XW = (bf16*)(smem + 270336);        // 8 waves x 16x32     (  8192 B)
  float* SC = (float*)(smem + 278528);       // 2 x 64 x 64 f32     ( 32768 B)

  const int tid   = threadIdx.x;
  const int lane  = tid & 31;
  const int wave  = tid >> 5;                // 0..7 (wave32)
  const int eh    = wave >> 2;               // e-half 0/1
  const int btile = wave & 3;                // 16-row b tile
  const int ht    = tid & 127;               // thread id within e-half

  const int t_x = 1 + step + s;              // embeddings time index
  const int t_c = s;                         // contexts time index

  // ---- stage X_s (64 x 1024 f32 -> bf16 LDS), all 256 threads ----
  for (int i = tid; i < 64 * 256; i += 256) {
    const int b  = i >> 8;
    const int d4 = i & 255;
    float4 v = *(const float4*)(emb + ((size_t)b * T_ + t_x) * DX_ + d4 * 4);
    bf16x4 o = { (bf16)v.x, (bf16)v.y, (bf16)v.z, (bf16)v.w };
    *(bf16x4*)(Xb + b * DX_ + d4 * 4) = o;
  }

  v8f sc0 = {0,0,0,0,0,0,0,0};
  v8f sc1 = sc0, sc2 = sc0, sc3 = sc0;       // 16b x 64c partial scores

  const float* WstepF = Wg + (size_t)step * DY_ * DX_;
  const bf16*  WstepB = Wb + (size_t)step * DY_ * DX_;
  const uint32_t WtLds = lds_addr32(Wt);     // LDS byte base of W tiles

  for (int it = 0; it < 16; ++it) {
    __syncthreads();                          // covers X staging + tile reuse
    const int e0 = 32 * (2 * it + eh);
    bf16* Wh = Wt + eh * 32 * DX_;

    if (ASYNCW) {
      // ---- async DMA: bf16 W rows [e0, e0+32) x 1024 -> LDS (no VGPR return)
#pragma unroll 4
      for (int k = 0; k < 32; ++k) {
        const int i  = ht + 128 * k;          // 4096 x 16B chunks per half
        const int er = i >> 7;                // row within tile (128 chunks/row)
        const int d8 = i & 127;
        const bf16* g = WstepB + (size_t)(e0 + er) * DX_ + d8 * 8;
        const uint32_t l = WtLds + (uint32_t)((eh * 32 + er) * 2048 + d8 * 16);
        async_ld_b128(l, g);
      }
    } else {
      // ---- fallback: f32 load + cvt + ds_store
#pragma unroll 4
      for (int k = 0; k < 64; ++k) {
        const int i  = ht + 128 * k;          // 8192 float4s per half
        const int er = i >> 8;
        const int d4 = i & 255;
        float4 v = *(const float4*)(WstepF + (size_t)(e0 + er) * DX_ + d4 * 4);
        bf16x4 o = { (bf16)v.x, (bf16)v.y, (bf16)v.z, (bf16)v.w };
        *(bf16x4*)(Wh + er * DX_ + d4 * 4) = o;
      }
    }

    // ---- stage C_s columns [e0, e0+32) for all 64 rows (f32 -> bf16)
    bf16* Ch = Ct + eh * 64 * 32;
#pragma unroll
    for (int k = 0; k < 4; ++k) {
      const int i  = ht + 128 * k;            // 512 float4s per half
      const int c  = i >> 3;
      const int e4 = i & 7;
      float4 v = *(const float4*)(ctx + ((size_t)c * T_ + t_c) * DY_ + e0 + e4 * 4);
      bf16x4 o = { (bf16)v.x, (bf16)v.y, (bf16)v.z, (bf16)v.w };
      *(bf16x4*)(Ch + c * 32 + e4 * 4) = o;
    }

    if (ASYNCW)
      asm volatile("s_wait_asynccnt 0" ::: "memory");  // LDS written by DMA
    __syncthreads();

    // ---- stage 1: xw (16b x 32e) = X[btile] @ W^T, K = 1024 ----
    v8f xw0 = {0,0,0,0,0,0,0,0};
    v8f xw1 = xw0;
#pragma unroll 4
    for (int d = 0; d < DX_; d += 32) {
      v16bf a  = load_a_frag(Xb, btile * 16, DX_, d);
      v16bf b0 = load_b_frag(Wh, 0,  DX_, d);   // e sub-tile 0
      v16bf b1 = load_b_frag(Wh, 16, DX_, d);   // e sub-tile 1
      xw0 = wmma_bf16(a, b0, xw0);
      xw1 = wmma_bf16(a, b1, xw1);
    }

    // ---- re-layout xw f32 accum -> bf16 A-tile via private LDS scratch ----
    bf16* xwT = XW + wave * (16 * 32);
    {
      const int col = lane & 15;
      const int rhi = (lane >> 4) * 8;
#pragma unroll
      for (int r = 0; r < 8; ++r) {
        xwT[(r + rhi) * 32 + col]      = (bf16)xw0[r];
        xwT[(r + rhi) * 32 + col + 16] = (bf16)xw1[r];
      }
    }
    asm volatile("s_wait_dscnt 0" ::: "memory");  // cross-lane LDS RAW in-wave

    // ---- stage 2: scores[btile, :] += xw @ C^T (K = 32 this iter) ----
    v16bf a2 = load_a_frag(xwT, 0, 32, 0);
    sc0 = wmma_bf16(a2, load_b_frag(Ch,  0, 32, 0), sc0);
    sc1 = wmma_bf16(a2, load_b_frag(Ch, 16, 32, 0), sc1);
    sc2 = wmma_bf16(a2, load_b_frag(Ch, 32, 32, 0), sc2);
    sc3 = wmma_bf16(a2, load_b_frag(Ch, 48, 32, 0), sc3);
  }

  // ---- dump partial scores (per e-half) to LDS ----
  {
    float* dst = SC + eh * 64 * 64;
    const int col = lane & 15;
    const int rhi = (lane >> 4) * 8;
#pragma unroll
    for (int r = 0; r < 8; ++r) {
      dst[(btile * 16 + r + rhi) * 64 +  0 + col] = sc0[r];
      dst[(btile * 16 + r + rhi) * 64 + 16 + col] = sc1[r];
      dst[(btile * 16 + r + rhi) * 64 + 32 + col] = sc2[r];
      dst[(btile * 16 + r + rhi) * 64 + 48 + col] = sc3[r];
    }
  }
  __syncthreads();

  // ---- epilogue: logsumexp over c, minus diagonal ----
  if (tid < 64) {
    const int b = tid;
    float m = -INFINITY;
    for (int c = 0; c < 64; ++c) {
      const float v = SC[b * 64 + c] + SC[4096 + b * 64 + c];
      m = fmaxf(m, v);
    }
    float sum = 0.f;
    for (int c = 0; c < 64; ++c) {
      const float v = SC[b * 64 + c] + SC[4096 + b * 64 + c];
      sum += expf(v - m);
    }
    const float pos = SC[b * 64 + b] + SC[4096 + b * 64 + b];
    const float lse = m + logf(sum) - logf((float)B_ * (float)S);
    const int off = 64 * (511 * step - (step * (step - 1)) / 2);
    out[off + b * S + s] = lse - pos;
  }
}

extern "C" void kernel_launch(void* const* d_in, const int* in_sizes, int n_in,
                              void* d_out, int out_size, void* d_ws, size_t ws_size,
                              hipStream_t stream) {
  (void)in_sizes; (void)n_in; (void)out_size;
  const float* emb = (const float*)d_in[0];   // (64, 512, 1024) f32
  const float* ctx = (const float*)d_in[1];   // (64, 512, 1024) f32
  const float* W   = (const float*)d_in[2];   // (5, 1024, 1024) f32
  float* out = (float*)d_out;                 // 162880 f32

  const size_t shmem  = 311296;               // 304 KB dynamic LDS (<= 320 KB/WGP)
  const size_t wbytes = (size_t)5 * DY_ * DX_ * sizeof(bf16);  // 10 MB bf16 W

  dim3 grid(511, 5);                          // s (max S=511) x step; blocks with s>=S exit
  dim3 block(256);                            // 8 waves (wave32)

  if (ws_size >= wbytes) {
    bf16* Wb = (bf16*)d_ws;
    // pre-pass: f32 -> bf16 W into workspace (5,242,880 elems, 8 per thread)
    convert_w_bf16_kernel<<<2560, 256, 0, stream>>>(W, Wb);
    hipFuncSetAttribute((const void*)infonce_fused_kernel<true>,
                        hipFuncAttributeMaxDynamicSharedMemorySize, (int)shmem);
    infonce_fused_kernel<true><<<grid, block, shmem, stream>>>(emb, ctx, W, Wb, out);
  } else {
    hipFuncSetAttribute((const void*)infonce_fused_kernel<false>,
                        hipFuncAttributeMaxDynamicSharedMemorySize, (int)shmem);
    infonce_fused_kernel<false><<<grid, block, shmem, stream>>>(emb, ctx, W, nullptr, out);
  }
}
